// PixelwiseViT_60000693125692
// MI455X (gfx1250) — matrix operand order repaired and checked by reading, not compiled, not run on hardware
//
#include <hip/hip_runtime.h>
#include <hip/hip_bf16.h>
#include <math.h>

// ---------------- problem constants ----------------
#define BB   16
#define CC   384
#define HH   28
#define WW   28
#define LL   (HH*WW)      // 784
#define FF   384
#define EE   384
#define FFN_ 1536
#define NB_  12
#define NH_  8
#define WS_  7
#define DH_  (FF/NH_)     // 48
#define NW_  (LL/WS_)     // 112
#define MM   (BB*LL)      // 12544

typedef __attribute__((ext_vector_type(16))) __bf16 v16bf;
typedef __attribute__((ext_vector_type(8)))  float  v8f;
typedef __attribute__((ext_vector_type(4)))  int    v4i;

enum { EPI_BIAS = 0, EPI_GELU = 1, EPI_RESID = 2, EPI_OUTTR = 3 };

// Async global->LDS copies (CDNA5 async path), feature-detected so the
// file compiles on toolchains without the builtins.
#if defined(__has_builtin)
#  if __has_builtin(__builtin_amdgcn_global_load_async_to_lds_b128) && \
      __has_builtin(__builtin_amdgcn_s_wait_asynccnt)
#    define USE_ASYNC_LDS 1
#  else
#    define USE_ASYNC_LDS 0
#  endif
#else
#  define USE_ASYNC_LDS 0
#endif

__device__ __forceinline__ unsigned short f2bf(float f) {
    unsigned int u = __float_as_uint(f);
    unsigned int r = u + 0x7FFFu + ((u >> 16) & 1u);   // round-to-nearest-even
    return (unsigned short)(r >> 16);
}
__device__ __forceinline__ float gelu_exact(float x) {
    return 0.5f * x * (1.0f + erff(x * 0.70710678118654752f));
}

#if USE_ASYNC_LDS
typedef __attribute__((address_space(1))) v4i gv4i;   // global int4
typedef __attribute__((address_space(3))) v4i lv4i;   // LDS int4
__device__ __forceinline__ void async_cp16(void* lds, const void* glob) {
    __builtin_amdgcn_global_load_async_to_lds_b128(
        (gv4i*)(void*)glob, (lv4i*)lds, 0, 0);
}
#endif

// ---------------------------------------------------------------------------
// bf16 WMMA GEMM:  Out = epilogue(A[M,K]bf16 @ WT[N,K]bf16^T + bias)
// 256 threads = 8 waves, tile 128x128, BK=32, DOUBLE-BUFFERED LDS.
// Wave (wm,wn) = (wave&1, wave>>1) owns a 64x32 region: 8 accumulators,
// 8 v_wmma_f32_16x16x32_bf16 per K-step fed by 12 ds_load_b128.
// Next tile is fetched (async-to-LDS when available) while WMMAs run on the
// current buffer -> one barrier per K-step, load latency off the critical path.
// All dims are exact tile multiples -> EXEC stays all-ones (WMMA requirement).
// ---------------------------------------------------------------------------
template <int EPI>
__global__ __launch_bounds__(256)
void gemm_bf16(const unsigned short* __restrict__ A,   // [M,K] bf16
               const unsigned short* __restrict__ WT,  // [N,K] bf16 (pre-transposed)
               const float* __restrict__ bias,
               const float* __restrict__ resid,
               const float* __restrict__ alphaPtr,
               void* __restrict__ OutV,
               int M, int N, int K)
{
    const int m0   = blockIdx.y * 128;
    const int n0   = blockIdx.x * 128;
    const int tid  = threadIdx.x;
    const int wave = tid >> 5;
    const int lane = tid & 31;
    const int wm   = wave & 1;          // 0..1 : 64-row half
    const int wn   = wave >> 1;         // 0..3 : 32-col quarter

    // two buffers: 128 rows x 32 K-halves, padded to 48 (96B rows, 16B aligned)
    __shared__ unsigned short sA[2][128][48];
    __shared__ unsigned short sB[2][128][48];

    v8f acc[4][2] = {};

    const int rowA = tid >> 2;          // 0..63 (plus +64 for second copy)
    const int k8   = (tid & 3) * 8;     // 8-half chunk within the 32-K tile

    const unsigned short* gA0 = A  + (size_t)(m0 + rowA)      * K + k8;
    const unsigned short* gA1 = A  + (size_t)(m0 + rowA + 64) * K + k8;
    const unsigned short* gB0 = WT + (size_t)(n0 + rowA)      * K + k8;
    const unsigned short* gB1 = WT + (size_t)(n0 + rowA + 64) * K + k8;

    // ---- prologue: fetch tile 0 into buffer 0 ----
#if USE_ASYNC_LDS
    async_cp16(&sA[0][rowA][k8],      gA0);
    async_cp16(&sA[0][rowA + 64][k8], gA1);
    async_cp16(&sB[0][rowA][k8],      gB0);
    async_cp16(&sB[0][rowA + 64][k8], gB1);
#else
    {
        uint4 a0 = *(const uint4*)gA0, a1 = *(const uint4*)gA1;
        uint4 b0 = *(const uint4*)gB0, b1 = *(const uint4*)gB1;
        *(uint4*)&sA[0][rowA][k8]      = a0;
        *(uint4*)&sA[0][rowA + 64][k8] = a1;
        *(uint4*)&sB[0][rowA][k8]      = b0;
        *(uint4*)&sB[0][rowA + 64][k8] = b1;
    }
#endif

    int buf = 0;
    for (int k0 = 0; k0 < K; k0 += 32) {
        const bool hasNext = (k0 + 32) < K;
        if (k0 + 64 < K) {              // prefetch tile-after-next into near cache
            __builtin_prefetch(gA0 + k0 + 64, 0, 3);
            __builtin_prefetch(gB0 + k0 + 64, 0, 3);
        }
#if USE_ASYNC_LDS
        __builtin_amdgcn_s_wait_asynccnt(0);
        __syncthreads();
        if (hasNext) {                  // fill the other buffer while we compute
            async_cp16(&sA[buf ^ 1][rowA][k8],      gA0 + k0 + 32);
            async_cp16(&sA[buf ^ 1][rowA + 64][k8], gA1 + k0 + 32);
            async_cp16(&sB[buf ^ 1][rowA][k8],      gB0 + k0 + 32);
            async_cp16(&sB[buf ^ 1][rowA + 64][k8], gB1 + k0 + 32);
        }
#else
        __syncthreads();
        uint4 na0 = {}, na1 = {}, nb0 = {}, nb1 = {};
        if (hasNext) {                  // issue next-tile loads early
            na0 = *(const uint4*)(gA0 + k0 + 32);
            na1 = *(const uint4*)(gA1 + k0 + 32);
            nb0 = *(const uint4*)(gB0 + k0 + 32);
            nb1 = *(const uint4*)(gB1 + k0 + 32);
        }
#endif
        // ---- fragments + 8 WMMAs on current buffer ----
        union Frag { v16bf v; uint4 q[2]; } fa[4], fb[2];
        const int kbA = (lane < 16) ? 0 : 8;   // A: halves {kbA..+7, kbA+16..+23}
        #pragma unroll
        for (int mi = 0; mi < 4; ++mi) {
            const int r = wm * 64 + mi * 16 + (lane & 15);
            fa[mi].q[0] = *(const uint4*)&sA[buf][r][kbA];
            fa[mi].q[1] = *(const uint4*)&sA[buf][r][kbA + 16];
        }
        const int kbB = (lane < 16) ? 0 : 8;   // B: halves {kbB..+7, kbB+16..+23}
        #pragma unroll
        for (int ni = 0; ni < 2; ++ni) {
            const int c = wn * 32 + ni * 16 + (lane & 15);
            fb[ni].q[0] = *(const uint4*)&sB[buf][c][kbB];
            fb[ni].q[1] = *(const uint4*)&sB[buf][c][kbB + 16];
        }
        #pragma unroll
        for (int mi = 0; mi < 4; ++mi)
            #pragma unroll
            for (int ni = 0; ni < 2; ++ni)
                acc[mi][ni] = __builtin_amdgcn_wmma_f32_16x16x32_bf16(
                    false, fa[mi].v, false, fb[ni].v, (short)0, acc[mi][ni], false, false);

#if !USE_ASYNC_LDS
        if (hasNext) {                  // commit next tile to the other buffer
            *(uint4*)&sA[buf ^ 1][rowA][k8]      = na0;
            *(uint4*)&sA[buf ^ 1][rowA + 64][k8] = na1;
            *(uint4*)&sB[buf ^ 1][rowA][k8]      = nb0;
            *(uint4*)&sB[buf ^ 1][rowA + 64][k8] = nb1;
        }
#endif
        buf ^= 1;
    }

    float alpha = 1.0f;
    if constexpr (EPI == EPI_RESID) alpha = alphaPtr[0];

    #pragma unroll
    for (int mi = 0; mi < 4; ++mi) {
        #pragma unroll
        for (int ni = 0; ni < 2; ++ni) {
            #pragma unroll
            for (int v = 0; v < 8; ++v) {
                int row = m0 + wm * 64 + mi * 16 + ((lane < 16) ? v : v + 8);
                int col = n0 + wn * 32 + ni * 16 + (lane & 15);
                float r = acc[mi][ni][v] + bias[col];
                if constexpr (EPI == EPI_GELU) {
                    ((unsigned short*)OutV)[(size_t)row * N + col] = f2bf(gelu_exact(r));
                } else if constexpr (EPI == EPI_RESID) {
                    ((float*)OutV)[(size_t)row * N + col] =
                        resid[(size_t)row * N + col] + alpha * r;
                } else if constexpr (EPI == EPI_OUTTR) {
                    int b = row / LL, l = row % LL;     // out[b, c, l]
                    ((float*)OutV)[((size_t)b * CC + col) * LL + l] = r;
                } else {
                    ((float*)OutV)[(size_t)row * N + col] = r;
                }
            }
        }
    }
}

// ---------------------------------------------------------------------------
// One-time weight convert + transpose: W[K,N] f32 -> WT[N,K] bf16
// ---------------------------------------------------------------------------
__global__ __launch_bounds__(256)
void convT(const float* __restrict__ W, unsigned short* __restrict__ WT, int K, int N)
{
    int idx = blockIdx.x * 256 + threadIdx.x;
    if (idx >= K * N) return;
    int k = idx / N, n = idx % N;
    WT[(size_t)n * K + k] = f2bf(W[idx]);
}

__global__ __launch_bounds__(256)
void conv_bf16(const float* __restrict__ X, unsigned short* __restrict__ Y, int n)
{
    int i = blockIdx.x * 256 + threadIdx.x;
    if (i < n) Y[i] = f2bf(X[i]);
}

// ---------------------------------------------------------------------------
// Fourier positional embedding + channel-last transpose -> bf16 [M, 768]
// ---------------------------------------------------------------------------
__global__ __launch_bounds__(256)
void build_xin(const float* __restrict__ x, const float* __restrict__ ew,
               const float* __restrict__ eb, unsigned short* __restrict__ Xin)
{
    int idx = blockIdx.x * blockDim.x + threadIdx.x;
    if (idx >= MM * (EE + CC)) return;
    int j = idx % (EE + CC);
    int m = idx / (EE + CC);
    int b = m / LL, l = m % LL;
    float val;
    if (j < EE) {
        float xn = 2.0f * (float)(l / WW) / (float)(WW - 1) - 1.0f;
        float yn = 2.0f * (float)(l % HH) / (float)(HH - 1) - 1.0f;
        val = sinf(xn * ew[j] + yn * ew[EE + j] + eb[j]);
    } else {
        val = x[((size_t)b * CC + (j - EE)) * LL + l];
    }
    Xin[(size_t)m * (EE + CC) + j] = f2bf(val);
}

// ---------------------------------------------------------------------------
// LayerNorm over F=384 (f32 in, bf16 out), one row per 128-thread block.
// ---------------------------------------------------------------------------
__global__ __launch_bounds__(128)
void layernorm_kernel(const float* __restrict__ X, const float* __restrict__ S,
                      const float* __restrict__ Bv, unsigned short* __restrict__ Y)
{
    const int row = blockIdx.x;
    const float* x = X + (size_t)row * FF;
    const int tid = threadIdx.x, wave = tid >> 5, lane = tid & 31;
    float v0 = x[tid], v1 = x[tid + 128], v2 = x[tid + 256];

    __shared__ float red[8];
    float s = v0 + v1 + v2;
    #pragma unroll
    for (int o = 16; o > 0; o >>= 1) s += __shfl_down(s, o, 32);
    if (lane == 0) red[wave] = s;
    __syncthreads();
    if (tid == 0) red[4] = (red[0] + red[1] + red[2] + red[3]) * (1.0f / FF);
    __syncthreads();
    const float mean = red[4];

    float d0 = v0 - mean, d1 = v1 - mean, d2 = v2 - mean;
    float q = d0 * d0 + d1 * d1 + d2 * d2;
    #pragma unroll
    for (int o = 16; o > 0; o >>= 1) q += __shfl_down(q, o, 32);
    __syncthreads();
    if (lane == 0) red[wave] = q;
    __syncthreads();
    if (tid == 0) red[5] = rsqrtf((red[0] + red[1] + red[2] + red[3]) * (1.0f / FF) + 1e-5f);
    __syncthreads();
    const float inv = red[5];

    unsigned short* y = Y + (size_t)row * FF;
    y[tid]       = f2bf(d0 * inv * S[tid]       + Bv[tid]);
    y[tid + 128] = f2bf(d1 * inv * S[tid + 128] + Bv[tid + 128]);
    y[tid + 256] = f2bf(d2 * inv * S[tid + 256] + Bv[tid + 256]);
}

// ---------------------------------------------------------------------------
// Windowed attention: one (batch, window, head) per 64-thread block.
// f32 qkv in, bf16 o out.  qkv layout: [M, 3F] (q | k | v).
// ---------------------------------------------------------------------------
__global__ __launch_bounds__(64)
void attn_kernel(const float* __restrict__ QKV, unsigned short* __restrict__ O)
{
    const int id   = blockIdx.x;               // b*NW*NH + w*NH + h
    const int head = id % NH_;
    const int w    = (id / NH_) % NW_;
    const int b    = id / (NH_ * NW_);
    const int rowBase = b * LL + w * WS_;
    const int qc = head * DH_, kc = FF + head * DH_, vc = 2 * FF + head * DH_;
    const float scale = 0.14433756729740643f;  // 1/sqrt(48)

    __shared__ float sq[WS_][DH_], sk[WS_][DH_], sv[WS_][DH_], att[WS_][WS_ + 1];

    for (int i = threadIdx.x; i < WS_ * DH_; i += 64) {
        int t = i / DH_, d = i % DH_;
        size_t roff = (size_t)(rowBase + t) * (3 * FF);
        sq[t][d] = QKV[roff + qc + d];
        sk[t][d] = QKV[roff + kc + d];
        sv[t][d] = QKV[roff + vc + d];
    }
    __syncthreads();

    for (int p = threadIdx.x; p < WS_ * WS_; p += 64) {
        int i = p / WS_, j = p % WS_;
        float s = 0.f;
        #pragma unroll
        for (int d = 0; d < DH_; ++d) s += sq[i][d] * sk[j][d];
        att[i][j] = s * scale;
    }
    __syncthreads();

    if (threadIdx.x < WS_) {
        int i = threadIdx.x;
        float mx = att[i][0];
        #pragma unroll
        for (int j = 1; j < WS_; ++j) mx = fmaxf(mx, att[i][j]);
        float sum = 0.f;
        #pragma unroll
        for (int j = 0; j < WS_; ++j) { float e = __expf(att[i][j] - mx); att[i][j] = e; sum += e; }
        float inv = 1.0f / sum;
        #pragma unroll
        for (int j = 0; j < WS_; ++j) att[i][j] *= inv;
    }
    __syncthreads();

    for (int p = threadIdx.x; p < WS_ * DH_; p += 64) {
        int i = p / DH_, d = p % DH_;
        float s = 0.f;
        #pragma unroll
        for (int j = 0; j < WS_; ++j) s += att[i][j] * sv[j][d];
        O[(size_t)(rowBase + i) * FF + head * DH_ + d] = f2bf(s);
    }
}

// ---------------------------------------------------------------------------
static inline void launch_gemm(int epi, const unsigned short* A, const unsigned short* WT,
                               const float* bias, const float* resid,
                               const float* alpha, void* out,
                               int M, int N, int K, hipStream_t s)
{
    dim3 g(N / 128, M / 128), blk(256);
    switch (epi) {
        case EPI_BIAS:  gemm_bf16<EPI_BIAS> <<<g, blk, 0, s>>>(A, WT, bias, resid, alpha, out, M, N, K); break;
        case EPI_GELU:  gemm_bf16<EPI_GELU> <<<g, blk, 0, s>>>(A, WT, bias, resid, alpha, out, M, N, K); break;
        case EPI_RESID: gemm_bf16<EPI_RESID><<<g, blk, 0, s>>>(A, WT, bias, resid, alpha, out, M, N, K); break;
        default:        gemm_bf16<EPI_OUTTR><<<g, blk, 0, s>>>(A, WT, bias, resid, alpha, out, M, N, K); break;
    }
}

static inline void launch_convT(const float* W, unsigned short* WT, int K, int N, hipStream_t s)
{
    int total = K * N;
    convT<<<(total + 255) / 256, 256, 0, s>>>(W, WT, K, N);
}

extern "C" void kernel_launch(void* const* d_in, const int* in_sizes, int n_in,
                              void* d_out, int out_size, void* d_ws, size_t ws_size,
                              hipStream_t stream)
{
    const float* x       = (const float*)d_in[0];
    const float* embed_w = (const float*)d_in[1];
    const float* embed_b = (const float*)d_in[2];
    const float* in_w    = (const float*)d_in[3];
    const float* in_b    = (const float*)d_in[4];
    const float* ln1_s   = (const float*)d_in[5];
    const float* ln1_b   = (const float*)d_in[6];
    const float* qkv_w   = (const float*)d_in[7];
    const float* qkv_b   = (const float*)d_in[8];
    const float* proj_w  = (const float*)d_in[9];
    const float* proj_b  = (const float*)d_in[10];
    const float* ln2_s   = (const float*)d_in[11];
    const float* ln2_b   = (const float*)d_in[12];
    const float* ffn_w1  = (const float*)d_in[13];
    const float* ffn_b1  = (const float*)d_in[14];
    const float* ffn_w2  = (const float*)d_in[15];
    const float* ffn_b2  = (const float*)d_in[16];
    const float* re_a    = (const float*)d_in[17];
    const float* out_w   = (const float*)d_in[18];
    const float* out_b   = (const float*)d_in[19];
    (void)in_sizes; (void)n_in; (void)out_size; (void)ws_size;

    // ---- workspace carve-up (all 256B aligned) ----
    char* ws = (char*)d_ws;
    const size_t S_H   = (size_t)MM * FF * 4;       // 19,267,584
    const size_t S_BIG = (size_t)MM * 3 * FF * 4;   // 57,802,752
    const size_t S_YB  = (size_t)MM * FF * 2;       //  9,633,792
    const size_t S_MID = (size_t)MM * FFN_ * 2;     // 38,535,168
    float*          h   = (float*)ws;
    float*          big = (float*)(ws + S_H);
    unsigned short* yb  = (unsigned short*)(ws + S_H + S_BIG);
    unsigned short* mid = (unsigned short*)(ws + S_H + S_BIG + S_YB);
    unsigned short* wbf = (unsigned short*)(ws + S_H + S_BIG + S_YB + S_MID);

    // bf16 transposed weight offsets (in halves)
    const size_t OFF_INW   = 0;
    const size_t SZ_INW    = (size_t)(EE + CC) * FF;                   // 294912
    const size_t SZ_QKV    = (size_t)FF * 3 * FF;                      // 442368
    const size_t SZ_PROJ   = (size_t)FF * FF;                          // 147456
    const size_t SZ_FFN    = (size_t)FF * FFN_;                        // 589824
    const size_t SZ_BLK    = SZ_QKV + SZ_PROJ + 2 * SZ_FFN;            // 1769472
    const size_t OFF_BLK0  = OFF_INW + SZ_INW;
    const size_t OFF_OUTW  = OFF_BLK0 + (size_t)NB_ * SZ_BLK;

    // ---- one-time weight convert + transpose to bf16 [N,K] ----
    launch_convT(in_w, wbf + OFF_INW, EE + CC, FF, stream);
    for (int blk = 0; blk < NB_; ++blk) {
        unsigned short* wb = wbf + OFF_BLK0 + (size_t)blk * SZ_BLK;
        launch_convT(qkv_w  + (size_t)blk * FF * 3 * FF, wb,                         FF,   3 * FF, stream);
        launch_convT(proj_w + (size_t)blk * FF * FF,     wb + SZ_QKV,                FF,   FF,     stream);
        launch_convT(ffn_w1 + (size_t)blk * FF * FFN_,   wb + SZ_QKV + SZ_PROJ,      FF,   FFN_,   stream);
        launch_convT(ffn_w2 + (size_t)blk * FFN_ * FF,   wb + SZ_QKV + SZ_PROJ + SZ_FFN, FFN_, FF, stream);
    }
    launch_convT(out_w, wbf + OFF_OUTW, FF, CC, stream);

    // ---- 1) positional embedding + concat -> mid[M, 768] (bf16) ----
    {
        int total = MM * (EE + CC);
        build_xin<<<(total + 255) / 256, 256, 0, stream>>>(x, embed_w, embed_b, mid);
    }
    // ---- 2) input projection -> h (f32) ----
    launch_gemm(EPI_BIAS, mid, wbf + OFF_INW, in_b, nullptr, nullptr, h, MM, FF, EE + CC, stream);

    // ---- 3) transformer blocks ----
    for (int blk = 0; blk < NB_; ++blk) {
        const float* alpha = re_a + blk;
        const unsigned short* wb = wbf + OFF_BLK0 + (size_t)blk * SZ_BLK;

        layernorm_kernel<<<MM, 128, 0, stream>>>(h, ln1_s + blk * FF, ln1_b + blk * FF, yb);
        launch_gemm(EPI_BIAS, yb, wb, qkv_b + (size_t)blk * 3 * FF,
                    nullptr, nullptr, big, MM, 3 * FF, FF, stream);
        attn_kernel<<<BB * NW_ * NH_, 64, 0, stream>>>(big, yb);
        launch_gemm(EPI_RESID, yb, wb + SZ_QKV, proj_b + (size_t)blk * FF,
                    h, alpha, h, MM, FF, FF, stream);
        layernorm_kernel<<<MM, 128, 0, stream>>>(h, ln2_s + blk * FF, ln2_b + blk * FF, yb);
        launch_gemm(EPI_GELU, yb, wb + SZ_QKV + SZ_PROJ, ffn_b1 + (size_t)blk * FFN_,
                    nullptr, nullptr, mid, MM, FFN_, FF, stream);
        launch_gemm(EPI_RESID, mid, wb + SZ_QKV + SZ_PROJ + SZ_FFN, ffn_b2 + (size_t)blk * FF,
                    h, alpha, h, MM, FF, FFN_, stream);
    }

    // ---- 4) output projection with [B,L,C] -> [B,C,H,W] transpose epilogue ----
    conv_bf16<<<(MM * FF + 255) / 256, 256, 0, stream>>>(h, yb, MM * FF);
    launch_gemm(EPI_OUTTR, yb, wbf + OFF_OUTW, out_b, nullptr, nullptr,
                (float*)d_out, MM, CC, FF, stream);
}
